// STN3d_Rotate_910533067692
// MI455X (gfx1250) — matrix operand order repaired and checked by reading, not compiled, not run on hardware
//
#include <hip/hip_runtime.h>
#include <hip/hip_bf16.h>

typedef _Float16 h8   __attribute__((ext_vector_type(8)));
typedef _Float16 v16h __attribute__((ext_vector_type(16)));
typedef float    v8f  __attribute__((ext_vector_type(8)));

#define P_TOTAL (32 * 4096)          // 131072 points
#define EPSF    1e-5f
#define PI_F    3.1415927410125732f

// ---- ordered-uint mapping so we can use integer atomicMax/Min for float max/min ----
__device__ __forceinline__ unsigned ordf(float f) {
  unsigned u = __float_as_uint(f);
  return (u & 0x80000000u) ? ~u : (u | 0x80000000u);
}
__device__ __forceinline__ float unordf(unsigned u) {
  return __uint_as_float((u & 0x80000000u) ? (u & 0x7FFFFFFFu) : ~u);
}

// ---- WMMA fragment loaders using contiguous K order (k = 16*h + s), same in A and B ----
// A (16x32 f16): lanes 0-15 rows M, 16 contiguous K values K{kh..kh+15}, kh=0/16
__device__ __forceinline__ v16h load_a_frag(const _Float16* __restrict__ W, int ldk,
                                            int mbase, int kbase, int lane) {
  int row = mbase + (lane & 15);
  int kh  = (lane < 16) ? 0 : 16;
  const _Float16* p = W + (size_t)row * ldk + kbase + kh;
  union { v16h v; h8 h[2]; } u;
  u.h[0] = *(const h8*)p;          // K = kbase+kh   .. +7
  u.h[1] = *(const h8*)(p + 8);    // K = kbase+kh+8 .. +15
  return u.v;
}
// B (32x16 f16): lane = column N, 16 contiguous K values (lo half lanes: K+0, hi: K+16)
__device__ __forceinline__ v16h load_b_frag(const _Float16* __restrict__ A, int ldk,
                                            int nbase, int kbase, int lane) {
  int col = nbase + (lane & 15);
  int kh  = (lane < 16) ? 0 : 16;
  return *(const v16h*)(A + (size_t)col * ldk + kbase + kh);
}
#define WMMA_F16(a, b, c) \
  __builtin_amdgcn_wmma_f32_16x16x32_f16(false, (a), false, (b), (short)0, (c), false, false)

// ---------------------------------------------------------------------------
// init: zero the float stats region, fill omin with 0xFFFFFFFF
__global__ void init_kernel(float* __restrict__ z, int nz, unsigned* __restrict__ mn, int nmin) {
  int t = blockIdx.x * 256 + threadIdx.x;
  if (t < nz)   z[t] = 0.0f;
  if (t < nmin) mn[t] = 0xFFFFFFFFu;
}

__global__ void cvt_kernel(const float* __restrict__ in, _Float16* __restrict__ out, int n) {
  int t = blockIdx.x * 256 + threadIdx.x;
  if (t < n) out[t] = (_Float16)in[t];
}

// conv1 (3->64): per point, pre-BN f16 out (point-major [p][64]) + channel stats
__global__ void conv1_kernel(const float* __restrict__ x, const float* __restrict__ w1,
                             const float* __restrict__ b1, _Float16* __restrict__ y1h,
                             float* __restrict__ gsum, float* __restrict__ gsq) {
  __shared__ float sw[192], sb[64], ssum[64], ssq[64];
  int tid = threadIdx.x;
  if (tid < 192) sw[tid] = w1[tid];
  if (tid < 64) { sb[tid] = b1[tid]; ssum[tid] = 0.f; ssq[tid] = 0.f; }
  __syncthreads();
  int p = blockIdx.x * 256 + tid;   // 0..131071
  int b = p >> 12, n = p & 4095;
  float x0 = x[(size_t)(b * 3 + 0) * 4096 + n];
  float x1 = x[(size_t)(b * 3 + 1) * 4096 + n];
  float x2 = x[(size_t)(b * 3 + 2) * 4096 + n];
  _Float16* out = y1h + (size_t)p * 64;
  #pragma unroll 8
  for (int o = 0; o < 64; ++o) {
    float v = fmaf(sw[o * 3 + 0], x0, fmaf(sw[o * 3 + 1], x1, fmaf(sw[o * 3 + 2], x2, sb[o])));
    out[o] = (_Float16)v;
    atomicAdd(&ssum[o], v);
    atomicAdd(&ssq[o], v * v);
  }
  __syncthreads();
  if (tid < 64) { atomicAdd(&gsum[tid], ssum[tid]); atomicAdd(&gsq[tid], ssq[tid]); }
}

// (sum, sumsq, gamma, beta) -> per-channel affine (scale, shift)
__global__ void finalize_kernel(const float* __restrict__ sum, const float* __restrict__ sq,
                                const float* __restrict__ g, const float* __restrict__ be,
                                float invcnt, int C,
                                float* __restrict__ scale, float* __restrict__ shift) {
  int c = blockIdx.x * 256 + threadIdx.x;
  if (c >= C) return;
  float m = sum[c] * invcnt;
  float v = sq[c] * invcnt - m * m;
  float s = g[c] * rsqrtf(v + EPSF);
  scale[c] = s;
  shift[c] = be[c] - m * s;
}

// apply BN+ReLU elementwise, f16 -> f16 (C is power of two, pass mask)
__global__ void bnrelu_kernel(const _Float16* __restrict__ y, const float* __restrict__ scale,
                              const float* __restrict__ shift, _Float16* __restrict__ a,
                              int cmask, int total) {
  int t = blockIdx.x * 256 + threadIdx.x;
  if (t >= total) return;
  int c = t & cmask;
  float v = (float)y[t] * scale[c] + shift[c];
  a[t] = (_Float16)fmaxf(v, 0.0f);
}

// conv2 (64->128) WMMA GEMM: out pre-BN f16 [p][128] + channel stats.
// 8 waves x Mtile16 cover all 128 channels; 256 points per block.
__global__ void conv2_gemm(const _Float16* __restrict__ w2h, const _Float16* __restrict__ act,
                           const float* __restrict__ bias, _Float16* __restrict__ yh,
                           float* __restrict__ gsum, float* __restrict__ gsq) {
  __shared__ float ssum[128], ssq[128];
  int tid = threadIdx.x, lane = tid & 31, wave = tid >> 5;
  if (tid < 128) { ssum[tid] = 0.f; ssq[tid] = 0.f; }
  __syncthreads();
  int mbase = wave * 16;
  v16h a0 = load_a_frag(w2h, 64, mbase, 0, lane);
  v16h a1 = load_a_frag(w2h, 64, mbase, 32, lane);
  int chbase = mbase + ((lane < 16) ? 0 : 8);
  float bl[8], lsum[8], lsq[8];
  #pragma unroll
  for (int r = 0; r < 8; ++r) { bl[r] = bias[chbase + r]; lsum[r] = 0.f; lsq[r] = 0.f; }
  int n0 = blockIdx.x * 256;
  for (int nt = 0; nt < 16; ++nt) {
    int nb = n0 + nt * 16;
    v16h b0 = load_b_frag(act, 64, nb, 0, lane);
    v16h b1 = load_b_frag(act, 64, nb, 32, lane);
    v8f c = {};
    c = WMMA_F16(a0, b0, c);
    c = WMMA_F16(a1, b1, c);
    int point = nb + (lane & 15);
    h8 hv;
    #pragma unroll
    for (int r = 0; r < 8; ++r) {
      float v = c[r] + bl[r];
      hv[r] = (_Float16)v;
      lsum[r] += v; lsq[r] += v * v;
    }
    *(h8*)(yh + (size_t)point * 128 + chbase) = hv;   // contiguous 16B store
  }
  #pragma unroll
  for (int r = 0; r < 8; ++r) { atomicAdd(&ssum[chbase + r], lsum[r]); atomicAdd(&ssq[chbase + r], lsq[r]); }
  __syncthreads();
  if (tid < 128) { atomicAdd(&gsum[tid], ssum[tid]); atomicAdd(&gsq[tid], ssq[tid]); }
}

// conv3 (128->1024) WMMA GEMM, fully fused epilogue: NO h3 store.
// Emits per-channel sum/sumsq and per-(b,channel) max/min of pre-BN values.
// grid = (512 nblocks of 256 points, 8 mblocks of 128 channels)
__global__ void conv3_gemm(const _Float16* __restrict__ w3h, const _Float16* __restrict__ act,
                           const float* __restrict__ bias,
                           float* __restrict__ gsum, float* __restrict__ gsq,
                           unsigned* __restrict__ gmax, unsigned* __restrict__ gmin) {
  __shared__ float ssum[128], ssq[128];
  __shared__ unsigned smax[128], smin[128];
  int tid = threadIdx.x, lane = tid & 31, wave = tid >> 5;
  if (tid < 128) { ssum[tid] = 0.f; ssq[tid] = 0.f; smax[tid] = 0u; smin[tid] = 0xFFFFFFFFu; }
  __syncthreads();
  int mb = blockIdx.y;                              // 0..7
  int mbase = wave * 16;                            // within the 128-ch block
  const _Float16* Wr = w3h + (size_t)mb * 128 * 128;
  v16h a0 = load_a_frag(Wr, 128, mbase, 0,  lane);
  v16h a1 = load_a_frag(Wr, 128, mbase, 32, lane);
  v16h a2 = load_a_frag(Wr, 128, mbase, 64, lane);
  v16h a3 = load_a_frag(Wr, 128, mbase, 96, lane);
  int chloc = mbase + ((lane < 16) ? 0 : 8);
  int chglob = mb * 128 + chloc;
  float bl[8], lsum[8], lsq[8], lmax[8], lmin[8];
  #pragma unroll
  for (int r = 0; r < 8; ++r) {
    bl[r] = bias[chglob + r];
    lsum[r] = 0.f; lsq[r] = 0.f;
    lmax[r] = -__builtin_inff(); lmin[r] = __builtin_inff();
  }
  int n0 = blockIdx.x * 256;                        // stays inside one b (4096 % 256 == 0)
  for (int nt = 0; nt < 16; ++nt) {
    int nb = n0 + nt * 16;
    v16h b0 = load_b_frag(act, 128, nb, 0,  lane);
    v16h b1 = load_b_frag(act, 128, nb, 32, lane);
    v16h b2 = load_b_frag(act, 128, nb, 64, lane);
    v16h b3 = load_b_frag(act, 128, nb, 96, lane);
    v8f c = {};
    c = WMMA_F16(a0, b0, c);
    c = WMMA_F16(a1, b1, c);
    c = WMMA_F16(a2, b2, c);
    c = WMMA_F16(a3, b3, c);
    #pragma unroll
    for (int r = 0; r < 8; ++r) {
      float v = c[r] + bl[r];
      lsum[r] += v; lsq[r] += v * v;
      lmax[r] = fmaxf(lmax[r], v); lmin[r] = fminf(lmin[r], v);
    }
  }
  #pragma unroll
  for (int r = 0; r < 8; ++r) {
    atomicAdd(&ssum[chloc + r], lsum[r]);
    atomicAdd(&ssq[chloc + r],  lsq[r]);
    atomicMax(&smax[chloc + r], ordf(lmax[r]));
    atomicMin(&smin[chloc + r], ordf(lmin[r]));
  }
  __syncthreads();
  if (tid < 128) {
    int ch = mb * 128 + tid;
    atomicAdd(&gsum[ch], ssum[tid]);
    atomicAdd(&gsq[ch],  ssq[tid]);
    int bidx = n0 >> 12;
    atomicMax(&gmax[(size_t)bidx * 1024 + ch], smax[tid]);
    atomicMin(&gmin[(size_t)bidx * 1024 + ch], smin[tid]);
  }
}

// pooled[b][o] = relu(bn(max_n h3)) — use min when BN scale < 0 (exact)
__global__ void pool_kernel(const unsigned* __restrict__ omax, const unsigned* __restrict__ omin,
                            const float* __restrict__ scale, const float* __restrict__ shift,
                            float* __restrict__ pooled) {
  int t = blockIdx.x * 256 + threadIdx.x;   // 0..32767
  int o = t & 1023;
  float s = scale[o];
  float x = (s >= 0.0f) ? unordf(omax[t]) : unordf(omin[t]);
  pooled[t] = fmaxf(0.0f, s * x + shift[o]);
}

// FC: out[b][o] = sum_k act(in[b][k]) * W[o][k] + bias[o]; optional BN-ReLU on input,
// optional batch stats of the (pre-BN) output. One thread per (b,o).
__global__ void fc_kernel(const float* __restrict__ in, const float* __restrict__ scale,
                          const float* __restrict__ shift, const float* __restrict__ W,
                          const float* __restrict__ bias, float* __restrict__ out,
                          float* __restrict__ osum, float* __restrict__ osq,
                          int K, int O, float postmul) {
  int t = blockIdx.x * blockDim.x + threadIdx.x;
  if (t >= 32 * O) return;
  int b = t / O, o = t - b * O;
  const float* row = in + (size_t)b * K;
  const float* wr  = W + (size_t)o * K;
  float acc = 0.0f;
  if (scale) {
    for (int k = 0; k < K; ++k) {
      float v = fmaxf(0.0f, row[k] * scale[k] + shift[k]);
      acc = fmaf(v, wr[k], acc);
    }
  } else {
    for (int k = 0; k < K; ++k) acc = fmaf(row[k], wr[k], acc);
  }
  float r = (acc + bias[o]) * postmul;
  out[t] = r;
  if (osum) { atomicAdd(&osum[o], r); atomicAdd(&osq[o], r * r); }
}

// Euler (ZYX composition Rz@Ry@Rx) -> 3x3 rotation, one thread per batch element
__global__ void rot_kernel(const float* __restrict__ ang, float* __restrict__ out) {
  int b = threadIdx.x;
  if (b >= 32) return;
  float a = ang[b * 3 + 0], bb = ang[b * 3 + 1], c = ang[b * 3 + 2];
  float sa, ca, sb, cb, sc, cc;
  sincosf(a, &sa, &ca);
  sincosf(bb, &sb, &cb);
  sincosf(c, &sc, &cc);
  float* o = out + b * 9;
  o[0] = cc * cb;  o[1] = cc * sb * sa - sc * ca;  o[2] = cc * sb * ca + sc * sa;
  o[3] = sc * cb;  o[4] = sc * sb * sa + cc * ca;  o[5] = sc * sb * ca - cc * sa;
  o[6] = -sb;      o[7] = cb * sa;                 o[8] = cb * ca;
}

extern "C" void kernel_launch(void* const* d_in, const int* in_sizes, int n_in,
                              void* d_out, int out_size, void* d_ws, size_t ws_size,
                              hipStream_t stream) {
  (void)in_sizes; (void)n_in; (void)out_size; (void)ws_size;
  const float* x   = (const float*)d_in[0];
  const float* w1  = (const float*)d_in[1];
  const float* b1  = (const float*)d_in[2];
  const float* g1  = (const float*)d_in[3];
  const float* be1 = (const float*)d_in[4];
  const float* w2  = (const float*)d_in[5];
  const float* b2  = (const float*)d_in[6];
  const float* g2  = (const float*)d_in[7];
  const float* be2 = (const float*)d_in[8];
  const float* w3  = (const float*)d_in[9];
  const float* b3  = (const float*)d_in[10];
  const float* g3  = (const float*)d_in[11];
  const float* be3 = (const float*)d_in[12];
  const float* fw1 = (const float*)d_in[13];
  const float* fb1 = (const float*)d_in[14];
  const float* g4  = (const float*)d_in[15];
  const float* be4 = (const float*)d_in[16];
  const float* fw2 = (const float*)d_in[17];
  const float* fb2 = (const float*)d_in[18];
  const float* g5  = (const float*)d_in[19];
  const float* be5 = (const float*)d_in[20];
  const float* fw3 = (const float*)d_in[21];
  const float* fb3 = (const float*)d_in[22];
  const float* g6  = (const float*)d_in[23];
  const float* be6 = (const float*)d_in[24];
  const float* fw4 = (const float*)d_in[25];
  const float* fb4 = (const float*)d_in[26];

  const size_t P = P_TOTAL;
  size_t off = 0;
  auto take = [&](size_t bytes) -> void* {
    void* p = (char*)d_ws + off;
    off = (off + bytes + 255) & ~(size_t)255;
    return p;
  };
  _Float16* y1h = (_Float16*)take(P * 64 * 2);
  _Float16* a1h = (_Float16*)take(P * 64 * 2);
  _Float16* y2h = (_Float16*)take(P * 128 * 2);
  _Float16* a2h = (_Float16*)take(P * 128 * 2);
  _Float16* w2h = (_Float16*)take(128 * 64 * 2);
  _Float16* w3h = (_Float16*)take(1024 * 128 * 2);
  // contiguous zero-init region: all Σ/Σ² accumulators then omax (bit-zero == valid init)
  float* statz = (float*)take((4224 + 32768) * 4);
  float *sum1 = statz,        *sq1 = statz + 64;
  float *sum2 = statz + 128,  *sq2 = statz + 256;
  float *sum3 = statz + 384,  *sq3 = statz + 1408;
  float *s4s  = statz + 2432, *s4q = statz + 2944;
  float *s5s  = statz + 3456, *s5q = statz + 3712;
  float *s6s  = statz + 3968, *s6q = statz + 4096;
  unsigned* omax = (unsigned*)(statz + 4224);
  unsigned* omin = (unsigned*)take(32768 * 4);
  float* ss = (float*)take(4224 * 4);
  float *scale1 = ss,        *shift1 = ss + 64;
  float *scale2 = ss + 128,  *shift2 = ss + 256;
  float *scale3 = ss + 384,  *shift3 = ss + 1408;
  float *scale4 = ss + 2432, *shift4 = ss + 2944;
  float *scale5 = ss + 3456, *shift5 = ss + 3712;
  float *scale6 = ss + 3968, *shift6 = ss + 4096;
  float* pooled = (float*)take(32768 * 4);
  float* z1 = (float*)take(32 * 512 * 4);
  float* z2 = (float*)take(32 * 256 * 4);
  float* z3 = (float*)take(32 * 128 * 4);
  float* angv = (float*)take(96 * 4);

  const float invPN = 1.0f / (float)P;
  const float invB  = 1.0f / 32.0f;

  init_kernel<<<145, 256, 0, stream>>>(statz, 4224 + 32768, omin, 32768);
  cvt_kernel<<<32, 256, 0, stream>>>(w2, w2h, 128 * 64);
  cvt_kernel<<<512, 256, 0, stream>>>(w3, w3h, 1024 * 128);

  conv1_kernel<<<512, 256, 0, stream>>>(x, w1, b1, y1h, sum1, sq1);
  finalize_kernel<<<1, 256, 0, stream>>>(sum1, sq1, g1, be1, invPN, 64, scale1, shift1);
  bnrelu_kernel<<<32768, 256, 0, stream>>>(y1h, scale1, shift1, a1h, 63, (int)(P * 64));

  conv2_gemm<<<512, 256, 0, stream>>>(w2h, a1h, b2, y2h, sum2, sq2);
  finalize_kernel<<<1, 256, 0, stream>>>(sum2, sq2, g2, be2, invPN, 128, scale2, shift2);
  bnrelu_kernel<<<65536, 256, 0, stream>>>(y2h, scale2, shift2, a2h, 127, (int)(P * 128));

  conv3_gemm<<<dim3(512, 8), 256, 0, stream>>>(w3h, a2h, b3, sum3, sq3, omax, omin);
  finalize_kernel<<<4, 256, 0, stream>>>(sum3, sq3, g3, be3, invPN, 1024, scale3, shift3);
  pool_kernel<<<128, 256, 0, stream>>>(omax, omin, scale3, shift3, pooled);

  fc_kernel<<<64, 256, 0, stream>>>(pooled, nullptr, nullptr, fw1, fb1, z1, s4s, s4q, 1024, 512, 1.0f);
  finalize_kernel<<<2, 256, 0, stream>>>(s4s, s4q, g4, be4, invB, 512, scale4, shift4);
  fc_kernel<<<32, 256, 0, stream>>>(z1, scale4, shift4, fw2, fb2, z2, s5s, s5q, 512, 256, 1.0f);
  finalize_kernel<<<1, 256, 0, stream>>>(s5s, s5q, g5, be5, invB, 256, scale5, shift5);
  fc_kernel<<<16, 256, 0, stream>>>(z2, scale5, shift5, fw3, fb3, z3, s6s, s6q, 256, 128, 1.0f);
  finalize_kernel<<<1, 128, 0, stream>>>(s6s, s6q, g6, be6, invB, 128, scale6, shift6);
  fc_kernel<<<1, 128, 0, stream>>>(z3, scale6, shift6, fw4, fb4, angv, nullptr, nullptr, 128, 3, PI_F);

  rot_kernel<<<1, 32, 0, stream>>>(angv, (float*)d_out);
}
